// ResidualQuantizer_30846455120248
// MI455X (gfx1250) — matrix operand order repaired and checked by reading, not compiled, not run on hardware
//
#include <hip/hip_runtime.h>

#define NQ 4
#define CK 2048
#define DIM 256
#define NB 16384

typedef __attribute__((ext_vector_type(16))) __bf16 v16bf;
typedef __attribute__((ext_vector_type(8)))  __bf16 v8bf;
typedef __attribute__((ext_vector_type(8)))  float  v8f;

// ---------------------------------------------------------------------------
// c_sq[q][k] = sum_d codebook[q][k][d]^2   (exact f32)
// ---------------------------------------------------------------------------
__global__ __launch_bounds__(256)
void rvq_csq_kernel(const float* __restrict__ cb, float* __restrict__ c_sq) {
    int t = blockIdx.x * 256 + threadIdx.x;      // 0 .. NQ*CK-1
    if (t >= NQ * CK) return;
    const float* row = cb + (size_t)t * DIM;
    float s = 0.f;
    #pragma unroll 8
    for (int d = 0; d < DIM; ++d) s += row[d] * row[d];
    c_sq[t] = s;
}

// ---------------------------------------------------------------------------
// Pack B = (-2 * codebook) as hi/lo bf16 in the WMMA B-matrix lane layout.
// Layout: [q][ntile(128)][kc(8)][lane(32)][16 bf16]
//   lane = g*16+n : column N = n, dword v holds K = g*16 + 2v, 2v+1 (within kc*32)
// ---------------------------------------------------------------------------
__global__ __launch_bounds__(256)
void rvq_packb_kernel(const float* __restrict__ cb,
                      __bf16* __restrict__ b_hi, __bf16* __restrict__ b_lo) {
    int q  = blockIdx.x >> 7;
    int nt = blockIdx.x & 127;
    for (int p = threadIdx.x; p < 2048; p += 256) {   // 2048 bf16-pairs per block
        int kc   = p >> 8;
        int rem  = p & 255;
        int lane = rem >> 3;
        int v    = rem & 7;
        int g = lane >> 4, n = lane & 15;
        int d0  = kc * 32 + g * 16 + 2 * v;
        int row = nt * 16 + n;
        const float* src = cb + (((size_t)q * CK + row) * DIM);
        float m0 = -2.0f * src[d0];
        float m1 = -2.0f * src[d0 + 1];
        __bf16 h0 = (__bf16)m0; __bf16 l0 = (__bf16)(m0 - (float)h0);
        __bf16 h1 = (__bf16)m1; __bf16 l1 = (__bf16)(m1 - (float)h1);
        size_t base = ((((size_t)q * 128 + nt) * 8 + kc) * 32 + lane) * 16 + 2 * v;
        b_hi[base] = h0; b_hi[base + 1] = h1;
        b_lo[base] = l0; b_lo[base + 1] = l1;
    }
}

// ---------------------------------------------------------------------------
// residual = inputs; A hi/lo bf16 split (row-major [b][d])
// ---------------------------------------------------------------------------
__global__ __launch_bounds__(256)
void rvq_packa_kernel(const float* __restrict__ inp, float* __restrict__ res,
                      __bf16* __restrict__ a_hi, __bf16* __restrict__ a_lo) {
    size_t total = (size_t)NB * DIM;
    for (size_t i = blockIdx.x * 256 + threadIdx.x; i < total; i += (size_t)gridDim.x * 256) {
        float x = inp[i];
        res[i] = x;
        __bf16 h = (__bf16)x;
        a_hi[i] = h;
        a_lo[i] = (__bf16)(x - (float)h);
    }
}

// ---------------------------------------------------------------------------
// Fused GEMM + argmin.  One block = one 16-row M-tile; 4 waves stripe the
// 128 N-tiles.  acc = c_sq - 2*r.c  via 3 bf16 WMMAs per 32-wide K chunk.
// ---------------------------------------------------------------------------
__global__ __launch_bounds__(128)
void rvq_gemm_argmin_kernel(const __bf16* __restrict__ a_hi,
                            const __bf16* __restrict__ a_lo,
                            const __bf16* __restrict__ b_hi,   // this stage
                            const __bf16* __restrict__ b_lo,   // this stage
                            const float*  __restrict__ c_sq,   // this stage [CK]
                            int* __restrict__ idxout)          // [NB]
{
    const int mt   = blockIdx.x;            // 0..1023
    const int wave = threadIdx.x >> 5;      // 0..3
    const int lane = threadIdx.x & 31;
    const int col  = lane & 15;
    const int g    = lane >> 4;

    // ---- hoist the 16x256 A tile (hi/lo) into registers, WMMA A layout ----
    const int row = mt * 16 + col;
    const __bf16* pah = a_hi + (size_t)row * DIM + g * 16;
    const __bf16* pal = a_lo + (size_t)row * DIM + g * 16;
    v16bf ah[8], al[8];
    #pragma unroll
    for (int kc = 0; kc < 8; ++kc) {
        v8bf h0 = *(const v8bf*)(pah + kc * 32);
        v8bf h1 = *(const v8bf*)(pah + kc * 32 + 8);
        ah[kc] = __builtin_shufflevector(h0, h1, 0,1,2,3,4,5,6,7,8,9,10,11,12,13,14,15);
        v8bf l0 = *(const v8bf*)(pal + kc * 32);
        v8bf l1 = *(const v8bf*)(pal + kc * 32 + 8);
        al[kc] = __builtin_shufflevector(l0, l1, 0,1,2,3,4,5,6,7,8,9,10,11,12,13,14,15);
    }

    float mv[8]; int mi[8];
    #pragma unroll
    for (int v = 0; v < 8; ++v) { mv[v] = __builtin_inff(); mi[v] = 0; }

    for (int nt = wave; nt < 128; nt += 4) {
        float cs = c_sq[nt * 16 + col];
        v8f acc;
        #pragma unroll
        for (int v = 0; v < 8; ++v) acc[v] = cs;

        const __bf16* pbh = b_hi + (size_t)nt * 4096 + lane * 16;
        const __bf16* pbl = b_lo + (size_t)nt * 4096 + lane * 16;
        #pragma unroll
        for (int kc = 0; kc < 8; ++kc) {
            v16bf bh = *(const v16bf*)(pbh + kc * 512);
            v16bf bl = *(const v16bf*)(pbl + kc * 512);
            acc = __builtin_amdgcn_wmma_f32_16x16x32_bf16(false, ah[kc], false, bh,
                                                          (short)0, acc, false, false);
            acc = __builtin_amdgcn_wmma_f32_16x16x32_bf16(false, ah[kc], false, bl,
                                                          (short)0, acc, false, false);
            acc = __builtin_amdgcn_wmma_f32_16x16x32_bf16(false, al[kc], false, bh,
                                                          (short)0, acc, false, false);
        }
        int n = nt * 16 + col;
        #pragma unroll
        for (int v = 0; v < 8; ++v)
            if (acc[v] < mv[v]) { mv[v] = acc[v]; mi[v] = n; }
    }

    // reduce across the 16 lanes sharing each row (xor masks stay in half-wave)
    #pragma unroll
    for (int off = 1; off < 16; off <<= 1) {
        #pragma unroll
        for (int v = 0; v < 8; ++v) {
            float ov = __shfl_xor(mv[v], off, 32);
            int   oi = __shfl_xor(mi[v], off, 32);
            if (ov < mv[v] || (ov == mv[v] && oi < mi[v])) { mv[v] = ov; mi[v] = oi; }
        }
    }

    __shared__ float sv[4][16];
    __shared__ int   si[4][16];
    if (lane == 0) {
        #pragma unroll
        for (int v = 0; v < 8; ++v) { sv[wave][v] = mv[v]; si[wave][v] = mi[v]; }
    } else if (lane == 16) {
        #pragma unroll
        for (int v = 0; v < 8; ++v) { sv[wave][8 + v] = mv[v]; si[wave][8 + v] = mi[v]; }
    }
    __syncthreads();
    if (threadIdx.x < 16) {
        float bv = sv[0][threadIdx.x]; int bi = si[0][threadIdx.x];
        #pragma unroll
        for (int w = 1; w < 4; ++w) {
            float ov = sv[w][threadIdx.x]; int oi = si[w][threadIdx.x];
            if (ov < bv || (ov == bv && oi < bi)) { bv = ov; bi = oi; }
        }
        idxout[mt * 16 + threadIdx.x] = bi;
    }
}

// ---------------------------------------------------------------------------
// residual -= codebook[q][idx]; refresh hi/lo A; emit stage index (as float)
// ---------------------------------------------------------------------------
__global__ __launch_bounds__(256)
void rvq_update_kernel(const float* __restrict__ cb, const int* __restrict__ idxbuf,
                       int q, float* __restrict__ res,
                       __bf16* __restrict__ a_hi, __bf16* __restrict__ a_lo,
                       float* __restrict__ out) {
    int b = blockIdx.x;
    int t = threadIdx.x;
    int idx = idxbuf[b];
    float c = cb[(((size_t)q * CK + idx) * DIM) + t];
    size_t i = (size_t)b * DIM + t;
    float r = res[i] - c;
    res[i] = r;
    __bf16 h = (__bf16)r;
    a_hi[i] = h;
    a_lo[i] = (__bf16)(r - (float)h);
    if (t == 0) out[(size_t)NB * DIM + (size_t)b * NQ + q] = (float)idx;
}

// ---------------------------------------------------------------------------
// quantized_out = inputs - residual_final
// ---------------------------------------------------------------------------
__global__ __launch_bounds__(256)
void rvq_final_kernel(const float* __restrict__ inp, const float* __restrict__ res,
                      float* __restrict__ out) {
    size_t i = (size_t)blockIdx.x * DIM + threadIdx.x;
    out[i] = inp[i] - res[i];
}

extern "C" void kernel_launch(void* const* d_in, const int* in_sizes, int n_in,
                              void* d_out, int out_size, void* d_ws, size_t ws_size,
                              hipStream_t stream) {
    const float* inp = (const float*)d_in[0];   // [NB, DIM] f32
    const float* cb  = (const float*)d_in[1];   // [NQ, CK, DIM] f32
    float* out = (float*)d_out;                 // NB*DIM quantized + NB*NQ indices

    char* w = (char*)d_ws;
    float*  res  = (float*)w;   w += (size_t)NB * DIM * sizeof(float);
    __bf16* a_hi = (__bf16*)w;  w += (size_t)NB * DIM * sizeof(__bf16);
    __bf16* a_lo = (__bf16*)w;  w += (size_t)NB * DIM * sizeof(__bf16);
    __bf16* b_hi = (__bf16*)w;  w += (size_t)NQ * CK * DIM * sizeof(__bf16);
    __bf16* b_lo = (__bf16*)w;  w += (size_t)NQ * CK * DIM * sizeof(__bf16);
    float*  c_sq = (float*)w;   w += (size_t)NQ * CK * sizeof(float);
    int*    idxb = (int*)w;

    rvq_csq_kernel  <<<(NQ * CK + 255) / 256, 256, 0, stream>>>(cb, c_sq);
    rvq_packb_kernel<<<NQ * 128, 256, 0, stream>>>(cb, b_hi, b_lo);
    rvq_packa_kernel<<<4096, 256, 0, stream>>>(inp, res, a_hi, a_lo);

    for (int q = 0; q < NQ; ++q) {
        rvq_gemm_argmin_kernel<<<NB / 16, 128, 0, stream>>>(
            a_hi, a_lo,
            b_hi + (size_t)q * CK * DIM,
            b_lo + (size_t)q * CK * DIM,
            c_sq + (size_t)q * CK,
            idxb);
        rvq_update_kernel<<<NB, 256, 0, stream>>>(cb, idxb, q, res, a_hi, a_lo, out);
    }
    rvq_final_kernel<<<NB, 256, 0, stream>>>(inp, res, out);
}
